// PC_MO_LSTM_5454608466712
// MI455X (gfx1250) — compile-verified, matrix operand order checked by reading, not run on hardware
//
#include <hip/hip_runtime.h>
#include <stdint.h>

// ---------------- model dims ----------------
#define BB 4
#define NN 8192
#define T_INF 3
#define NUM_PREDF 2
#define C1 64
#define C2 128
#define C3 256
#define M1 512
#define M2 256
#define M3 128

// ---------------- WMMA GEMM tiling ----------------
// Block tile 128x64, 8 waves in a 4x2 grid, each wave owns a 2x2 grid of
// 16x16 f32 accumulators (wave tile 32x32) -> 4 v_wmma per wave per K chunk.
#define GBM 128
#define GBN 64
#define GBK 32
#define AS_STRIDE 40  // halfs; multiple of 8 so rows stay 16B aligned

typedef __attribute__((ext_vector_type(16))) _Float16 v16h;
typedef __attribute__((ext_vector_type(8)))  _Float16 v8h;
typedef __attribute__((ext_vector_type(8)))  float    v8f;

__device__ __forceinline__ float sigf(float x) { return 1.0f / (1.0f + expf(-x)); }

// =========================================================================
// GEMM: C[M,N] = A[M,K] * W[K,N] (+bias) (+=C if accum) (ReLU if relu)
// f32 in/out, f16 WMMA inner product (v_wmma_f32_16x16x32_f16), LDS double
// buffered so global staging of chunk k+1 overlaps WMMA of chunk k.
// =========================================================================
__global__ __launch_bounds__(256) void k_gemm(
    const float* __restrict__ A, const float* __restrict__ W,
    const float* __restrict__ bias, float* __restrict__ Cm,
    int M, int N, int K, int relu, int accum)
{
  __shared__ __align__(16) _Float16 As[2][GBM * AS_STRIDE];
  __shared__ __align__(16) _Float16 Bs[2][GBN * AS_STRIDE];

  const int t    = threadIdx.x;
  const int wave = t >> 5;
  const int lane = t & 31;
  const int wm   = wave >> 1;       // 0..3 -> rows wm*32
  const int wn   = wave & 1;        // 0..1 -> cols wn*32
  const int m0   = blockIdx.x * GBM;
  const int n0   = blockIdx.y * GBN;
  const int lh   = lane >> 4;       // half-wave select
  const int l16  = lane & 15;

  const int nk = (K + GBK - 1) / GBK;

  // ---- stage one K chunk into LDS buffer `buf` ----
  auto stage = [&](int buf, int k0) {
    // A tile: 128x32 f32 -> f16, 4 float-quads per thread
    if (m0 + GBM <= M && k0 + GBK <= K) {
      #pragma unroll
      for (int i = 0; i < 4; ++i) {
        int f  = t + i * 256;        // quad id
        int r  = f >> 3;
        int c4 = (f & 7) << 2;
        const float* p = A + (size_t)(m0 + r) * K + (k0 + c4);
        float x0 = p[0], x1 = p[1], x2 = p[2], x3 = p[3];
        _Float16* d = &As[buf][r * AS_STRIDE + c4];
        d[0] = (_Float16)x0; d[1] = (_Float16)x1;
        d[2] = (_Float16)x2; d[3] = (_Float16)x3;
      }
    } else {
      #pragma unroll
      for (int i = 0; i < 4; ++i) {
        int f  = t + i * 256;
        int r  = f >> 3;
        int c4 = (f & 7) << 2;
        int gm = m0 + r, gk = k0 + c4;
        float x0 = 0.f, x1 = 0.f, x2 = 0.f, x3 = 0.f;
        if (gm < M) {
          const float* p = A + (size_t)gm * K + gk;
          if (gk + 3 < K) { x0 = p[0]; x1 = p[1]; x2 = p[2]; x3 = p[3]; }
          else {
            if (gk     < K) x0 = p[0];
            if (gk + 1 < K) x1 = p[1];
            if (gk + 2 < K) x2 = p[2];
          }
        }
        _Float16* d = &As[buf][r * AS_STRIDE + c4];
        d[0] = (_Float16)x0; d[1] = (_Float16)x1;
        d[2] = (_Float16)x2; d[3] = (_Float16)x3;
      }
    }
    // W tile transposed: Bs[n][k] = W[k0+k][n0+n], 8 elements per thread
    if (k0 + GBK <= K && n0 + GBN <= N) {
      #pragma unroll
      for (int i = 0; i < 8; ++i) {
        int e  = t + i * 256;
        int kk = e >> 6;
        int nn = e & 63;
        Bs[buf][nn * AS_STRIDE + kk] =
            (_Float16)W[(size_t)(k0 + kk) * N + (n0 + nn)];
      }
    } else {
      #pragma unroll
      for (int i = 0; i < 8; ++i) {
        int e  = t + i * 256;
        int kk = e >> 6;
        int nn = e & 63;
        int gk = k0 + kk, gn = n0 + nn;
        float x = (gk < K && gn < N) ? W[(size_t)gk * N + gn] : 0.0f;
        Bs[buf][nn * AS_STRIDE + kk] = (_Float16)x;
      }
    }
  };

  v8f acc[2][2] = {};

  stage(0, 0);
  for (int kt = 0; kt < nk; ++kt) {
    __syncthreads();                 // staged chunk kt now visible
    if (kt + 1 < nk) stage((kt + 1) & 1, (kt + 1) * GBK);
    const int buf = kt & 1;

    // fragments per CDNA5 16-bit layouts: lane (m=l16, h=lh),
    // halfs [0..7] = K in [h*8, h*8+8), halfs [8..15] = K in [16+h*8, ...)
    v16h af[2], bf[2];
    #pragma unroll
    for (int i2 = 0; i2 < 2; ++i2) {
      const int arow = wm * 32 + i2 * 16 + l16;
      v8h a0 = *(const v8h*)&As[buf][arow * AS_STRIDE + lh * 8];
      v8h a1 = *(const v8h*)&As[buf][arow * AS_STRIDE + 16 + lh * 8];
      const int brow = wn * 32 + i2 * 16 + l16;
      v8h b0 = *(const v8h*)&Bs[buf][brow * AS_STRIDE + lh * 8];
      v8h b1 = *(const v8h*)&Bs[buf][brow * AS_STRIDE + 16 + lh * 8];
      #pragma unroll
      for (int i = 0; i < 8; ++i) {
        af[i2][i] = a0[i]; af[i2][8 + i] = a1[i];
        bf[i2][i] = b0[i]; bf[i2][8 + i] = b1[i];
      }
    }
    #pragma unroll
    for (int ti = 0; ti < 2; ++ti)
      #pragma unroll
      for (int tj = 0; tj < 2; ++tj)
        acc[ti][tj] = __builtin_amdgcn_wmma_f32_16x16x32_f16(
            false, af[ti], false, bf[tj], (short)0, acc[ti][tj], false, false);
  }

  // ---- write back: C/D layout, VGPR r -> M = r + lh*8, N = l16 ----
  #pragma unroll
  for (int ti = 0; ti < 2; ++ti) {
    #pragma unroll
    for (int tj = 0; tj < 2; ++tj) {
      #pragma unroll
      for (int r = 0; r < 8; ++r) {
        int mo = m0 + wm * 32 + ti * 16 + lh * 8 + r;
        int no = n0 + wn * 32 + tj * 16 + l16;
        if (mo < M && no < N) {
          size_t off = (size_t)mo * N + no;
          float v = acc[ti][tj][r];
          if (bias)  v += bias[no];
          if (accum) v += Cm[off];
          if (relu && v < 0.f) v = 0.f;
          Cm[off] = v;
        }
      }
    }
  }
}

// =========================================================================
// kNN: per query keep top-k (k<=32) nearest of n source points.
// Source points staged through LDS with CDNA5 async global->LDS copies.
// =========================================================================
#define KNN_TILE 128
__global__ __launch_bounds__(256) void k_knn(
    const float* __restrict__ q, const float* __restrict__ s,
    int M, int n, int k, int* __restrict__ idx)
{
  __shared__ __align__(16) float sp[KNN_TILE * 3];
  const int b  = blockIdx.y;
  const int qi = blockIdx.x * 256 + threadIdx.x;

  float qx = 0.f, qy = 0.f, qz = 0.f;
  if (qi < M) {
    const float* qp = q + ((size_t)b * M + qi) * 3;
    qx = qp[0]; qy = qp[1]; qz = qp[2];
  }
  float bd[32]; int bi[32];
  #pragma unroll
  for (int j = 0; j < 32; ++j) { bd[j] = 3.0e30f; bi[j] = 0; }

  const float* sb = s + (size_t)b * n * 3;
  for (int t0 = 0; t0 < n; t0 += KNN_TILE) {
    int cnt = n - t0; if (cnt > KNN_TILE) cnt = KNN_TILE;
    int nw  = cnt * 3;
    for (int e = threadIdx.x; e < nw; e += 256) {
#if defined(__gfx1250__)
      unsigned lo = (unsigned)(unsigned long long)&sp[e];
      const float* gp = sb + (size_t)t0 * 3 + e;
      asm volatile("global_load_async_to_lds_b32 %0, %1, off"
                   :: "v"(lo), "v"(gp) : "memory");
#else
      sp[e] = sb[(size_t)t0 * 3 + e];
#endif
    }
#if defined(__gfx1250__)
    asm volatile("s_wait_asynccnt 0" ::: "memory");
#endif
    __syncthreads();
    if (qi < M) {
      for (int j = 0; j < cnt; ++j) {
        float dx = qx - sp[j * 3 + 0];
        float dy = qy - sp[j * 3 + 1];
        float dz = qz - sp[j * 3 + 2];
        float d  = dx * dx + dy * dy + dz * dz;
        if (d < bd[k - 1]) {
          int p = k - 1;
          while (p > 0 && bd[p - 1] > d) { bd[p] = bd[p - 1]; bi[p] = bi[p - 1]; --p; }
          bd[p] = d; bi[p] = t0 + j;
        }
      }
    }
    __syncthreads();
  }
  if (qi < M)
    for (int j = 0; j < k; ++j) idx[((size_t)b * M + qi) * k + j] = bi[j];
}

// ---------------- farthest point sampling (1 block / batch) ----------------
__global__ __launch_bounds__(256) void k_fps(
    const float* __restrict__ xyz, int n, int m,
    int* __restrict__ out, float* __restrict__ dist)
{
  const int b = blockIdx.x;
  const float* p = xyz + (size_t)b * n * 3;
  float* db = dist + (size_t)b * n;
  __shared__ float smax[256];
  __shared__ int   simax[256];
  __shared__ int   s_far;
  const int t = threadIdx.x;
  for (int i = t; i < n; i += 256) db[i] = 1.0e30f;
  if (t == 0) s_far = 0;
  __syncthreads();
  for (int it = 0; it < m; ++it) {
    int far = s_far;
    if (t == 0) out[(size_t)b * m + it] = far;
    float fx = p[far * 3 + 0], fy = p[far * 3 + 1], fz = p[far * 3 + 2];
    float lm = -1.f; int li = 0;
    for (int i = t; i < n; i += 256) {
      float dx = p[i * 3 + 0] - fx, dy = p[i * 3 + 1] - fy, dz = p[i * 3 + 2] - fz;
      float d = dx * dx + dy * dy + dz * dz;
      float o = db[i];
      d = d < o ? d : o;
      db[i] = d;
      if (d > lm) { lm = d; li = i; }
    }
    smax[t] = lm; simax[t] = li;
    __syncthreads();
    for (int s2 = 128; s2 > 0; s2 >>= 1) {
      if (t < s2 && smax[t + s2] > smax[t]) { smax[t] = smax[t + s2]; simax[t] = simax[t + s2]; }
      __syncthreads();
    }
    if (t == 0) s_far = simax[0];
    __syncthreads();
  }
}

// ---------------- small glue kernels ----------------
__global__ void k_transpose_b3n(const float* __restrict__ in, float* __restrict__ out, int n) {
  int b = blockIdx.y;
  size_t e = (size_t)blockIdx.x * 256 + threadIdx.x;
  if (e >= (size_t)n * 3) return;
  int i = (int)(e / 3), c = (int)(e % 3);
  out[((size_t)b * n + i) * 3 + c] = in[((size_t)b * 3 + c) * n + i];
}

__global__ void k_gather_rows(const float* __restrict__ src, const int* __restrict__ idx,
                              float* __restrict__ out, int n, int M, int C) {
  int b = blockIdx.y;
  size_t e = (size_t)blockIdx.x * 256 + threadIdx.x;
  if (e >= (size_t)M * C) return;
  int row = (int)(e / C), c = (int)(e % C);
  int s = idx[(size_t)b * M + row];
  out[((size_t)b * M + row) * C + c] = src[((size_t)b * n + s) * C + c];
}

__global__ void k_group_sa(const float* __restrict__ f, const float* __restrict__ xyz,
                           const float* __restrict__ centers, const int* __restrict__ nidx,
                           float* __restrict__ out, int n, int m, int k, int Cin) {
  int b = blockIdx.y;
  int Co = Cin + 3;
  size_t e = (size_t)blockIdx.x * 256 + threadIdx.x;
  if (e >= (size_t)m * k * Co) return;
  int row = (int)(e / Co), c = (int)(e % Co);
  int i = row / k, j = row % k;
  int s = nidx[((size_t)b * m + i) * k + j];
  float v;
  if (c < Cin) v = f[((size_t)b * n + s) * Cin + c];
  else {
    int cc = c - Cin;
    v = xyz[((size_t)b * n + s) * 3 + cc] - centers[((size_t)b * m + i) * 3 + cc];
  }
  out[(((size_t)b * m + i) * k + j) * Co + c] = v;
}

__global__ void k_maxpool_k(const float* __restrict__ in, float* __restrict__ out,
                            int rows, int k, int C) {
  size_t e = (size_t)blockIdx.x * 256 + threadIdx.x;
  if (e >= (size_t)rows * C) return;
  size_t row = e / C; int c = (int)(e % C);
  float mx = -3.0e30f;
  for (int j = 0; j < k; ++j) {
    float v = in[(row * k + j) * C + c];
    mx = v > mx ? v : mx;
  }
  out[row * C + c] = mx;
}

__global__ void k_gather_nb(const float* __restrict__ src, const int* __restrict__ nidx,
                            float* __restrict__ out, int m, int msrc, int k, int C) {
  int b = blockIdx.y;
  size_t e = (size_t)blockIdx.x * 256 + threadIdx.x;
  if (e >= (size_t)m * k * C) return;
  int row = (int)(e / C), c = (int)(e % C);
  int i = row / k, j = row % k;
  int s = nidx[((size_t)b * m + i) * k + j];
  out[(((size_t)b * m + i) * k + j) * C + c] = src[((size_t)b * msrc + s) * C + c];
}

__global__ void k_build_rel(const float* __restrict__ x_i, const float* __restrict__ x_last,
                            const int* __restrict__ nidx, float dt, float* __restrict__ out,
                            int m, int msrc, int k) {
  int b = blockIdx.y;
  size_t e = (size_t)blockIdx.x * 256 + threadIdx.x;
  if (e >= (size_t)m * k * 4) return;
  int row = (int)(e / 4), c = (int)(e % 4);
  int i = row / k, j = row % k;
  int s = nidx[((size_t)b * m + i) * k + j];
  float v = (c < 3)
      ? x_i[((size_t)b * m + i) * 3 + c] - x_last[((size_t)b * msrc + s) * 3 + c]
      : dt;
  out[(((size_t)b * m + i) * k + j) * 4 + c] = v;
}

__global__ void k_qkp(const float* __restrict__ q, const float* __restrict__ kf,
                      const float* __restrict__ pos, float* __restrict__ out,
                      size_t Rk, int k, int C) {
  size_t e = (size_t)blockIdx.x * 256 + threadIdx.x;
  if (e >= Rk * (size_t)C) return;
  size_t r = e / C; int c = (int)(e % C);
  out[e] = q[(r / k) * C + c] - kf[e] + pos[e];
}

__global__ void k_attn(const float* __restrict__ a, const float* __restrict__ vf,
                       const float* __restrict__ pos, float* __restrict__ out,
                       size_t Rq, int k, int C) {
  size_t e = (size_t)blockIdx.x * 256 + threadIdx.x;
  if (e >= Rq * (size_t)C) return;
  size_t r = e / C; int c = (int)(e % C);
  size_t base = (r * k) * (size_t)C + c;
  float mx = -3.0e30f;
  for (int j = 0; j < k; ++j) { float v = a[base + (size_t)j * C]; mx = v > mx ? v : mx; }
  float s = 0.f, acc = 0.f;
  for (int j = 0; j < k; ++j) {
    size_t o = base + (size_t)j * C;
    float w = expf(a[o] - mx);
    s += w;
    acc += w * (vf[o] + pos[o]);
  }
  out[e] = acc / s;
}

__global__ void k_lstm_cell(const float* __restrict__ g, float* __restrict__ h,
                            float* __restrict__ c, int R, int H) {
  size_t e = (size_t)blockIdx.x * 256 + threadIdx.x;
  if (e >= (size_t)R * H) return;
  size_t r = e / H; int j = (int)(e % H);
  size_t base = r * (size_t)(4 * H);
  float ig = sigf(g[base + j]);
  float fg = sigf(g[base + H + j]);
  float gg = tanhf(g[base + 2 * H + j]);
  float og = sigf(g[base + 3 * H + j]);
  float cn = fg * c[e] + ig * gg;
  c[e] = cn;
  h[e] = og * tanhf(cn);
}

__global__ void k_zero(float* __restrict__ p, size_t n) {
  size_t e = (size_t)blockIdx.x * 256 + threadIdx.x;
  if (e < n) p[e] = 0.f;
}
__global__ void k_copy(const float* __restrict__ a, float* __restrict__ b, size_t n) {
  size_t e = (size_t)blockIdx.x * 256 + threadIdx.x;
  if (e < n) b[e] = a[e];
}

__global__ void k_concat(const float* __restrict__ a, const float* __restrict__ b,
                         float* __restrict__ out, int R, int Ca, int Cb) {
  size_t e = (size_t)blockIdx.x * 256 + threadIdx.x;
  int Co = Ca + Cb;
  if (e >= (size_t)R * Co) return;
  size_t r = e / Co; int c = (int)(e % Co);
  out[e] = (c < Ca) ? a[r * Ca + c] : b[r * Cb + (c - Ca)];
}

__global__ void k_fp_interp(const float* __restrict__ xc, const float* __restrict__ pc,
                            const float* __restrict__ pf, const int* __restrict__ nidx,
                            float* __restrict__ out, int Mf, int Mc, int k, int C) {
  int b = blockIdx.y;
  size_t e = (size_t)blockIdx.x * 256 + threadIdx.x;
  if (e >= (size_t)Mf * C) return;
  int row = (int)(e / C), c = (int)(e % C);
  const float* qr = pf + ((size_t)b * Mf + row) * 3;
  float ws = 0.f, acc = 0.f;
  for (int j = 0; j < k; ++j) {
    int s = nidx[((size_t)b * Mf + row) * k + j];
    const float* pp = pc + ((size_t)b * Mc + s) * 3;
    float dx = qr[0] - pp[0], dy = qr[1] - pp[1], dz = qr[2] - pp[2];
    float w = 1.0f / (dx * dx + dy * dy + dz * dz + 1e-8f);
    ws += w;
    acc += w * xc[((size_t)b * Mc + s) * C + c];
  }
  out[((size_t)b * Mf + row) * C + c] = acc / ws;
}

__global__ void k_add_out(const float* __restrict__ base, const float* __restrict__ h,
                          float* __restrict__ pcnext, float* __restrict__ dout, size_t n) {
  size_t e = (size_t)blockIdx.x * 256 + threadIdx.x;
  if (e >= n) return;
  float v = base[e] + h[e];
  pcnext[e] = v;
  dout[e]   = v;
}

// =========================================================================
// Host-side orchestration
// =========================================================================
namespace {

struct Arena {
  char* base; size_t off;
  void reset() { off = 0; }
  void* ab(size_t bytes) { void* r = base + off; off = (off + bytes + 255) & ~(size_t)255; return r; }
  float* f(size_t n)  { return (float*)ab(n * 4); }
  int*   ip(size_t n) { return (int*)ab(n * 4); }
};

struct Enc { float *f1, *x1, *f2, *x2, *f3, *x3; };

// Param pointer order (flattened make_params insertion order, after input_xyz, num_pred):
enum ParamIdx {
  SA1 = 0, SA2 = 6, SA3 = 12,
  LPT1 = 18, LPT2 = 29, LPT3 = 40,          // Wq,Wk,Wv,Wp1,bp1,Wp2,bp2,Wa1,ba1,Wa2,ba2
  LSTM1 = 51, LSTM2 = 57, LSTM3 = 63,       // (Wih,Whh,b) x2
  FP32P = 69, FP21P = 73, FP10P = 77,       // (W,b) x2
  CLSP = 81                                  // 4 weights
};

inline dim3 g1(size_t n) { return dim3((unsigned)((n + 255) / 256)); }
inline dim3 gb(size_t per_batch) { return dim3((unsigned)((per_batch + 255) / 256), BB); }

void gemm(hipStream_t st, const float* A, const float* W, const float* bias,
          float* C, long M, long N, long K, int relu, int accum) {
  dim3 gr((unsigned)((M + GBM - 1) / GBM), (unsigned)((N + GBN - 1) / GBN));
  k_gemm<<<gr, 256, 0, st>>>(A, W, bias, C, (int)M, (int)N, (int)K, relu, accum);
}

void sa_mod(hipStream_t st, Arena& scr, const float* f, const float* xyz,
            int n, int m, int k, int Cin, const float* const* L,
            const int* dims, float* fout, float* xout) {
  scr.reset();
  int*   cidx = scr.ip((size_t)BB * m);
  float* dist = scr.f((size_t)BB * n);
  k_fps<<<dim3(BB), 256, 0, st>>>(xyz, n, m, cidx, dist);
  k_gather_rows<<<gb((size_t)m * 3), 256, 0, st>>>(xyz, cidx, xout, n, m, 3);
  int* nidx = scr.ip((size_t)BB * m * k);
  k_knn<<<dim3((unsigned)((m + 255) / 256), BB), 256, 0, st>>>(xout, xyz, m, n, k, nidx);
  int w0 = Cin + 3;
  size_t R = (size_t)BB * m * k;
  float* g0 = scr.f(R * w0);
  k_group_sa<<<gb((size_t)m * k * w0), 256, 0, st>>>(f, xyz, xout, nidx, g0, n, m, k, Cin);
  int maxd = dims[0] > dims[1] ? dims[0] : dims[1];
  if (dims[2] > maxd) maxd = dims[2];
  float* t0 = scr.f(R * maxd);
  float* t1 = scr.f(R * maxd);
  const float* h = g0; int cin = w0;
  float* outs[3] = { t0, t1, t0 };
  for (int l = 0; l < 3; ++l) {
    gemm(st, h, L[2 * l], L[2 * l + 1], outs[l], (long)R, dims[l], cin, 1, 0);
    h = outs[l]; cin = dims[l];
  }
  k_maxpool_k<<<g1((size_t)BB * m * cin), 256, 0, st>>>(h, fout, BB * m, k, cin);
}

void lpt_mod(hipStream_t st, Arena& scr, const float* f_i, const float* f_last,
             const float* x_i, const float* x_last, int m, int msrc, int k, int C,
             float dt, const float* const* p, float* out) {
  scr.reset();
  size_t Rq = (size_t)BB * m, Rk = Rq * k;
  int* nidx = scr.ip(Rk);
  k_knn<<<dim3((unsigned)((m + 255) / 256), BB), 256, 0, st>>>(x_i, x_last, m, msrc, k, nidx);
  float* q  = scr.f(Rq * C);
  gemm(st, f_i, p[0], nullptr, q, (long)Rq, C, C, 0, 0);
  float* ka = scr.f((size_t)BB * msrc * C);
  gemm(st, f_last, p[1], nullptr, ka, (long)BB * msrc, C, C, 0, 0);
  float* va = scr.f((size_t)BB * msrc * C);
  gemm(st, f_last, p[2], nullptr, va, (long)BB * msrc, C, C, 0, 0);
  float* kf = scr.f(Rk * C);
  k_gather_nb<<<gb((size_t)m * k * C), 256, 0, st>>>(ka, nidx, kf, m, msrc, k, C);
  float* vf = scr.f(Rk * C);
  k_gather_nb<<<gb((size_t)m * k * C), 256, 0, st>>>(va, nidx, vf, m, msrc, k, C);
  float* rel = scr.f(Rk * 4);
  k_build_rel<<<gb((size_t)m * k * 4), 256, 0, st>>>(x_i, x_last, nidx, dt, rel, m, msrc, k);
  float* t0 = scr.f(Rk * C);
  gemm(st, rel, p[3], p[4], t0, (long)Rk, C, 4, 1, 0);
  float* pos = scr.f(Rk * C);
  gemm(st, t0, p[5], p[6], pos, (long)Rk, C, C, 0, 0);
  k_qkp<<<g1(Rk * C), 256, 0, st>>>(q, kf, pos, t0, Rk, k, C);
  float* a1 = scr.f(Rk * C);
  gemm(st, t0, p[7], p[8], a1, (long)Rk, C, C, 1, 0);
  gemm(st, a1, p[9], p[10], kf, (long)Rk, C, C, 0, 0);   // reuse kf as attention logits
  k_attn<<<g1(Rq * C), 256, 0, st>>>(kf, vf, pos, out, Rq, k, C);
}

void lstm_last(hipStream_t st, Arena& scr, float* const* seq, int T, int R,
               int Cin, int H, const float* const* p, float* out) {
  scr.reset();
  float* hs[T_INF + NUM_PREDF];
  for (int t = 0; t < T; ++t) hs[t] = scr.f((size_t)R * H);
  float* h = scr.f((size_t)R * H);
  float* c = scr.f((size_t)R * H);
  float* g = scr.f((size_t)R * 4 * H);
  const float* xs[T_INF + NUM_PREDF];
  for (int t = 0; t < T; ++t) xs[t] = seq[t];
  int cin = Cin;
  for (int l = 0; l < 2; ++l) {
    k_zero<<<g1((size_t)R * H), 256, 0, st>>>(h, (size_t)R * H);
    k_zero<<<g1((size_t)R * H), 256, 0, st>>>(c, (size_t)R * H);
    for (int t = 0; t < T; ++t) {
      gemm(st, xs[t], p[l * 3 + 0], p[l * 3 + 2], g, R, 4 * H, cin, 0, 0);
      gemm(st, h,     p[l * 3 + 1], nullptr,      g, R, 4 * H, H,   0, 1);
      k_lstm_cell<<<g1((size_t)R * H), 256, 0, st>>>(g, h, c, R, H);
      k_copy<<<g1((size_t)R * H), 256, 0, st>>>(h, hs[t], (size_t)R * H);
    }
    for (int t = 0; t < T; ++t) xs[t] = hs[t];
    cin = H;
  }
  k_copy<<<g1((size_t)R * H), 256, 0, st>>>(h, out, (size_t)R * H);
}

void fp_mod(hipStream_t st, Arena& scr, const float* xc, const float* pc,
            const float* xskip, int Cskip, const float* pf,
            int Mc, int Mf, int k, int Cc, const float* const* L,
            int d0, int d1, float* out) {
  scr.reset();
  int* nidx = scr.ip((size_t)BB * Mf * k);
  k_knn<<<dim3((unsigned)((Mf + 255) / 256), BB), 256, 0, st>>>(pf, pc, Mf, Mc, k, nidx);
  float* xi = scr.f((size_t)BB * Mf * Cc);
  k_fp_interp<<<gb((size_t)Mf * Cc), 256, 0, st>>>(xc, pc, pf, nidx, xi, Mf, Mc, k, Cc);
  const float* h = xi; int cin = Cc;
  if (xskip) {
    float* cc = scr.f((size_t)BB * Mf * (Cc + Cskip));
    k_concat<<<g1((size_t)BB * Mf * (Cc + Cskip)), 256, 0, st>>>(xi, xskip, cc, BB * Mf, Cc, Cskip);
    h = cc; cin = Cc + Cskip;
  }
  float* t0 = scr.f((size_t)BB * Mf * d0);
  gemm(st, h,  L[0], L[1], t0,  (long)BB * Mf, d0, cin, 1, 0);
  gemm(st, t0, L[2], L[3], out, (long)BB * Mf, d1, d0,  1, 0);
}

void decode(hipStream_t st, Arena& pers, Arena& scr,
            const float* tm1, const float* tm2, const float* tm3,
            const float* x1, const float* x2, const float* x3,
            const float* last_xyz, const float* const* P,
            float* pcnext, float* dout_slot) {
  float* h2 = pers.f((size_t)BB * M2 * 576);
  fp_mod(st, scr, tm3, x3, tm2, 3 * C2, x2, M3, M2, 8, 3 * C3, P + FP32P, 768, 576, h2);
  float* h1 = pers.f((size_t)BB * M1 * 384);
  fp_mod(st, scr, h2, x2, tm1, 3 * C1, x1, M2, M1, 16, 576, P + FP21P, 576, 384, h1);
  float* h0 = pers.f((size_t)BB * NN * 128);
  fp_mod(st, scr, h1, x1, nullptr, 0, last_xyz, M1, NN, 32, 384, P + FP10P, 256, 128, h0);
  static const int cd[5] = { 128, 128, 64, 16, 3 };
  float* ca = pers.f((size_t)BB * NN * 128);
  float* cb = pers.f((size_t)BB * NN * 64);
  const float* h = h0;
  float* outs[4] = { ca, cb, ca, cb };
  for (int l = 0; l < 4; ++l) {
    gemm(st, h, P[CLSP + l], nullptr, outs[l], (long)BB * NN, cd[l + 1], cd[l], 0, 0);
    h = outs[l];
  }
  k_add_out<<<g1((size_t)BB * NN * 3), 256, 0, st>>>(last_xyz, h, pcnext, dout_slot,
                                                     (size_t)BB * NN * 3);
}

} // namespace

extern "C" void kernel_launch(void* const* d_in, const int* in_sizes, int n_in,
                              void* d_out, int out_size, void* d_ws, size_t ws_size,
                              hipStream_t stream) {
  (void)in_sizes; (void)n_in; (void)out_size;
  const float* input_xyz = (const float*)d_in[0];       // [T_IN][B][3][N]
  const float* const* P  = (const float* const*)(d_in + 2);

  Arena pers{ (char*)d_ws, 0 };
  Arena scr { (char*)d_ws + ws_size / 2, 0 };

  // frames: [B][3][N] -> [B][N][3]
  float* fx[T_INF + NUM_PREDF];
  for (int t = 0; t < T_INF; ++t) {
    fx[t] = pers.f((size_t)BB * NN * 3);
    k_transpose_b3n<<<gb((size_t)NN * 3), 256, 0, stream>>>(
        input_xyz + (size_t)t * BB * 3 * NN, fx[t], NN);
  }

  static const int d1[3] = { 32, 32, C1 };
  static const int d2[3] = { 96, 96, C2 };
  static const int d3[3] = { 128, 128, C3 };

  Enc E[T_INF + NUM_PREDF];
  auto do_encode = [&](const float* pc, Enc& e) {
    e.x1 = pers.f((size_t)BB * M1 * 3); e.f1 = pers.f((size_t)BB * M1 * C1);
    e.x2 = pers.f((size_t)BB * M2 * 3); e.f2 = pers.f((size_t)BB * M2 * C2);
    e.x3 = pers.f((size_t)BB * M3 * 3); e.f3 = pers.f((size_t)BB * M3 * C3);
    sa_mod(stream, scr, pc,   pc,   NN, M1, 32, 3,  P + SA1, d1, e.f1, e.x1);
    sa_mod(stream, scr, e.f1, e.x1, M1, M2, 16, C1, P + SA2, d2, e.f2, e.x2);
    sa_mod(stream, scr, e.f2, e.x2, M2, M3, 8,  C2, P + SA3, d3, e.f3, e.x3);
  };
  for (int t = 0; t < T_INF; ++t) do_encode(fx[t], E[t]);
  const float* last_xyz = fx[T_INF - 1];

  const float dt = 1.0f / (float)(T_INF + NUM_PREDF);

  float* ft1[8]; float* ft2[8]; float* ft3[8]; int nft = 0;
  float* ftt1 = pers.f((size_t)BB * M1 * C1);
  float* ftt2 = pers.f((size_t)BB * M2 * C2);
  float* ftt3 = pers.f((size_t)BB * M3 * C3);

  for (int i = 0; i < T_INF; ++i) {
    if (i < T_INF - 1) {
      ft1[nft] = pers.f((size_t)BB * M1 * C1);
      ft2[nft] = pers.f((size_t)BB * M2 * C2);
      ft3[nft] = pers.f((size_t)BB * M3 * C3);
      lpt_mod(stream, scr, E[i+1].f1, E[i].f1, E[i+1].x1, E[i].x1, M1, M1, 32, C1, dt, P + LPT1, ft1[nft]);
      lpt_mod(stream, scr, E[i+1].f2, E[i].f2, E[i+1].x2, E[i].x2, M2, M2, 16, C2, dt, P + LPT2, ft2[nft]);
      lpt_mod(stream, scr, E[i+1].f3, E[i].f3, E[i+1].x3, E[i].x3, M3, M3, 8,  C3, dt, P + LPT3, ft3[nft]);
      ++nft;
    } else {
      lpt_mod(stream, scr, E[i].f1, E[i].f1, E[i].x1, E[i].x1, M1, M1, 32, C1, 0.f, P + LPT1, ftt1);
      lpt_mod(stream, scr, E[i].f2, E[i].f2, E[i].x2, E[i].x2, M2, M2, 16, C2, 0.f, P + LPT2, ftt2);
      lpt_mod(stream, scr, E[i].f3, E[i].f3, E[i].x3, E[i].x3, M3, M3, 8,  C3, 0.f, P + LPT3, ftt3);
    }
  }

  float* pcnext = pers.f((size_t)BB * NN * 3);
  int cur = T_INF - 1;

  for (int pidx = 0; pidx < NUM_PREDF; ++pidx) {
    if (pidx > 0) {
      do_encode(pcnext, E[cur + 1]);
      ++cur;
      ft1[nft] = pers.f((size_t)BB * M1 * C1);
      ft2[nft] = pers.f((size_t)BB * M2 * C2);
      ft3[nft] = pers.f((size_t)BB * M3 * C3);
      lpt_mod(stream, scr, E[cur].f1, E[cur-1].f1, E[cur].x1, E[cur-1].x1, M1, M1, 32, C1, dt, P + LPT1, ft1[nft]);
      lpt_mod(stream, scr, E[cur].f2, E[cur-1].f2, E[cur].x2, E[cur-1].x2, M2, M2, 16, C2, dt, P + LPT2, ft2[nft]);
      lpt_mod(stream, scr, E[cur].f3, E[cur-1].f3, E[cur].x3, E[cur-1].x3, M3, M3, 8,  C3, dt, P + LPT3, ft3[nft]);
      ++nft;
      lpt_mod(stream, scr, E[cur].f1, E[cur].f1, E[cur].x1, E[cur].x1, M1, M1, 32, C1, 0.f, P + LPT1, ftt1);
      lpt_mod(stream, scr, E[cur].f2, E[cur].f2, E[cur].x2, E[cur].x2, M2, M2, 16, C2, 0.f, P + LPT2, ftt2);
      lpt_mod(stream, scr, E[cur].f3, E[cur].f3, E[cur].x3, E[cur].x3, M3, M3, 8,  C3, 0.f, P + LPT3, ftt3);
    }
    // tm = concat(lstm_last(ft_list), ftt)
    float* lo1 = pers.f((size_t)BB * M1 * 2 * C1);
    lstm_last(stream, scr, ft1, nft, BB * M1, C1, 2 * C1, P + LSTM1, lo1);
    float* tm1 = pers.f((size_t)BB * M1 * 3 * C1);
    k_concat<<<g1((size_t)BB * M1 * 3 * C1), 256, 0, stream>>>(lo1, ftt1, tm1, BB * M1, 2 * C1, C1);

    float* lo2 = pers.f((size_t)BB * M2 * 2 * C2);
    lstm_last(stream, scr, ft2, nft, BB * M2, C2, 2 * C2, P + LSTM2, lo2);
    float* tm2 = pers.f((size_t)BB * M2 * 3 * C2);
    k_concat<<<g1((size_t)BB * M2 * 3 * C2), 256, 0, stream>>>(lo2, ftt2, tm2, BB * M2, 2 * C2, C2);

    float* lo3 = pers.f((size_t)BB * M3 * 2 * C3);
    lstm_last(stream, scr, ft3, nft, BB * M3, C3, 2 * C3, P + LSTM3, lo3);
    float* tm3 = pers.f((size_t)BB * M3 * 3 * C3);
    k_concat<<<g1((size_t)BB * M3 * 3 * C3), 256, 0, stream>>>(lo3, ftt3, tm3, BB * M3, 2 * C3, C3);

    const Enc& ec = E[cur];
    decode(stream, pers, scr, tm1, tm2, tm3, ec.x1, ec.x2, ec.x3, last_xyz, P,
           pcnext, (float*)d_out + (size_t)pidx * BB * NN * 3);
  }
}
